// Model_30846455120585
// MI455X (gfx1250) — matrix operand order repaired and checked by reading, not compiled, not run on hardware
//
#include <hip/hip_runtime.h>
#include <hip/hip_bf16.h>
#include <math.h>

// Problem constants (match reference)
#define N_NODES 50000
#define EDGES   600000
#define FDIM    256
#define HDIM    128
#define QDIM    64
#define RREL    2
#define LDEP    2

typedef __attribute__((ext_vector_type(16))) __bf16 bf16x16;
typedef __attribute__((ext_vector_type(8)))  __bf16 bf16x8;
typedef __attribute__((ext_vector_type(4)))  __bf16 bf16x4;
typedef __attribute__((ext_vector_type(8)))  float  f32x8;

#define LDS_STRIDE 40   // bf16 elems per staged row: 32 data + 8 pad, keeps 16B alignment

// ---------------------------------------------------------------------------
// Generic bf16-WMMA GEMM:  C[M x NT*16] = act( rowScale(A)[M x K] @ B + bias )
//   A: f32 row-major (lda).  B: f32; TRANSB=0 -> B is (K x Nc, ldb),
//   TRANSB=1 -> B is (Nc x K, ldb) i.e. C = A @ B^T.
//   NT / TRANSB / HAS_SCALE are template params: accumulators stay in VGPRs,
//   staging loops are branch-free (full-tile fast path; only the single M-tail
//   block takes the guarded path).  K multiple of 32.
//   accum=1: C += A@B (existing C preloaded as WMMA accumulator).
// Block: 256 threads = 8 waves; block tile 128 rows; each wave owns 16 rows
// and all NT column tiles.  All NT B fragments are loaded before the WMMA
// chain so the DS loads clause up and the NT WMMAs issue back-to-back.
// ---------------------------------------------------------------------------
template <int NT, int TRANSB, int HAS_SCALE>
__global__ void __launch_bounds__(256)
k_gemm_bf16(const float* __restrict__ A, int lda,
            const float* __restrict__ B, int ldb,
            const float* __restrict__ rowScale,
            const float* __restrict__ bias,
            float* __restrict__ C, int ldc,
            int M, int K, int doTanh, int accum)
{
    constexpr int Nc = NT * 16;
    __shared__ __bf16 As[128 * LDS_STRIDE];   // [row][k]   (10240 B)
    __shared__ __bf16 Bs[Nc  * LDS_STRIDE];   // [col][k]

    const int tid  = threadIdx.x;
    const int lane = tid & 31;
    const int wv   = tid >> 5;
    const int rowBase = blockIdx.x * 128;
    const int half = lane >> 4;        // 0: lanes 0-15, 1: lanes 16-31
    const int rc   = lane & 15;
    const bool full = (rowBase + 128 <= M);

    f32x8 acc[NT];
    const f32x8 vzero = {0.f,0.f,0.f,0.f,0.f,0.f,0.f,0.f};
    #pragma unroll
    for (int ct = 0; ct < NT; ++ct) acc[ct] = vzero;

    if (accum) {
        #pragma unroll
        for (int ct = 0; ct < NT; ++ct) {
            const int col = ct * 16 + rc;
            #pragma unroll
            for (int g = 0; g < 8; ++g) {
                const int row = rowBase + wv * 16 + half * 8 + g;
                acc[ct][g] = (full || row < M) ? C[(size_t)row * ldc + col] : 0.0f;
            }
        }
    }

    for (int k0 = 0; k0 < K; k0 += 32) {
        // --- stage A tile (128 x 32) f32 -> bf16, fused per-row scale ---
        if (full) {
            #pragma unroll
            for (int ii = 0; ii < 4; ++ii) {             // 1024 float4 chunks / 256 thr
                const int i  = tid + ii * 256;
                const int rr = i >> 3;
                const int cc = (i & 7) << 2;
                const int grow = rowBase + rr;
                const float4 v = *(const float4*)&A[(size_t)grow * lda + k0 + cc];
                const float sc = HAS_SCALE ? rowScale[grow] : 1.0f;
                bf16x4 b;
                b[0] = (__bf16)(v.x * sc);
                b[1] = (__bf16)(v.y * sc);
                b[2] = (__bf16)(v.z * sc);
                b[3] = (__bf16)(v.w * sc);
                *(bf16x4*)&As[rr * LDS_STRIDE + cc] = b;
            }
        } else {
            #pragma unroll
            for (int ii = 0; ii < 4; ++ii) {
                const int i  = tid + ii * 256;
                const int rr = i >> 3;
                const int cc = (i & 7) << 2;
                const int grow = rowBase + rr;
                float4 v = make_float4(0.f, 0.f, 0.f, 0.f);
                float sc = 1.0f;
                if (grow < M) {
                    v = *(const float4*)&A[(size_t)grow * lda + k0 + cc];
                    if (HAS_SCALE) sc = rowScale[grow];
                }
                bf16x4 b;
                b[0] = (__bf16)(v.x * sc);
                b[1] = (__bf16)(v.y * sc);
                b[2] = (__bf16)(v.z * sc);
                b[3] = (__bf16)(v.w * sc);
                *(bf16x4*)&As[rr * LDS_STRIDE + cc] = b;
            }
        }
        // --- stage B tile (Nc x 32) into [col][k] layout, branch-free ---
        if (TRANSB) {
            // B rows (length K) contiguous along k: float4 along k, bf16x4 store
            constexpr int CH = (Nc * 32) / 4;            // float4 chunks
            #pragma unroll
            for (int ii = 0; ii < CH / 256; ++ii) {
                const int i  = tid + ii * 256;
                const int n  = i >> 3;
                const int kk = (i & 7) << 2;
                const float4 v = *(const float4*)&B[(size_t)n * ldb + k0 + kk];
                bf16x4 b;
                b[0] = (__bf16)v.x; b[1] = (__bf16)v.y;
                b[2] = (__bf16)v.z; b[3] = (__bf16)v.w;
                *(bf16x4*)&Bs[n * LDS_STRIDE + kk] = b;
            }
        } else {
            // B rows (length Nc) contiguous along n: float4 along n, 4 strided stores
            constexpr int CH = (Nc * 32) / 4;
            constexpr int NPR = Nc / 4;                  // float4 chunks per k-row
            #pragma unroll
            for (int ii = 0; ii < CH / 256; ++ii) {
                const int i  = tid + ii * 256;
                const int kk = i / NPR;
                const int nb = (i % NPR) * 4;
                const float4 v = *(const float4*)&B[(size_t)(k0 + kk) * ldb + nb];
                Bs[(nb + 0) * LDS_STRIDE + kk] = (__bf16)v.x;
                Bs[(nb + 1) * LDS_STRIDE + kk] = (__bf16)v.y;
                Bs[(nb + 2) * LDS_STRIDE + kk] = (__bf16)v.z;
                Bs[(nb + 3) * LDS_STRIDE + kk] = (__bf16)v.w;
            }
        }
        __syncthreads();

        // A fragment (ISA 16-bit A 16x32 layout):
        // lane<16: row=rc, K = {0..7} then {16..23}; lane>=16: K = {8..15},{24..31}
        const __bf16* pa = &As[(wv * 16 + rc) * LDS_STRIDE + half * 8];
        const bf16x8 alo = *(const bf16x8*)pa;
        const bf16x8 ahi = *(const bf16x8*)(pa + 16);
        bf16x16 af;
        #pragma unroll
        for (int i = 0; i < 8; ++i) { af[i] = alo[i]; af[i + 8] = ahi[i]; }

        // Load ALL B fragments first (DS loads clause up, single wait), then
        // issue the NT WMMAs back-to-back.
        bf16x16 bfr[NT];
        #pragma unroll
        for (int ct = 0; ct < NT; ++ct) {
            // B fragment: lane<16 -> col=rc, K = {0..7},{16..23} ; lane>=16 -> K = {8..15},{24..31}
            const __bf16* pb = &Bs[(ct * 16 + rc) * LDS_STRIDE + half * 8];
            const bf16x8 b0 = *(const bf16x8*)pb;
            const bf16x8 b1 = *(const bf16x8*)(pb + 16);
            #pragma unroll
            for (int i = 0; i < 8; ++i) { bfr[ct][i] = b0[i]; bfr[ct][i + 8] = b1[i]; }
        }
        #pragma unroll
        for (int ct = 0; ct < NT; ++ct) {
            acc[ct] = __builtin_amdgcn_wmma_f32_16x16x32_bf16(
                false, af, false, bfr[ct], (short)0, acc[ct], false, false);
        }
        __syncthreads();
    }

    // epilogue (C/D layout: VGPR g -> row g + 8*half, col = rc)
    #pragma unroll
    for (int ct = 0; ct < NT; ++ct) {
        const int col = ct * 16 + rc;
        const float bv = (bias && !accum) ? bias[col] : 0.0f;
        #pragma unroll
        for (int g = 0; g < 8; ++g) {
            const int row = rowBase + wv * 16 + half * 8 + g;
            if (full || row < M) {
                float v = acc[ct][g] + bv;
                if (doTanh) v = tanhf(v);
                C[(size_t)row * ldc + col] = v;
            }
        }
    }
}

// ---------------------------------------------------------------------------
// Degree histogram + rsqrt(clamp(deg,1))
// ---------------------------------------------------------------------------
__global__ void k_degree(const int* __restrict__ src, const int* __restrict__ dst,
                         float* __restrict__ degO, float* __restrict__ degI, int E)
{
    const int e = blockIdx.x * blockDim.x + threadIdx.x;
    if (e < E) {
        atomicAdd(&degO[src[e]], 1.0f);
        atomicAdd(&degI[dst[e]], 1.0f);
    }
}

__global__ void k_rsqrt_clamp(float* __restrict__ p, int n)
{
    const int i = blockIdx.x * blockDim.x + threadIdx.x;
    if (i < n) p[i] = rsqrtf(fmaxf(p[i], 1.0f));
}

// ---------------------------------------------------------------------------
// Edge gather/scatter-add: agg[dst] += m[src].  One wave per edge
// (512B coalesced gather, scalar-broadcast index loads, fp32 atomics in L2).
// ---------------------------------------------------------------------------
__launch_bounds__(256)
__global__ void k_scatter(const float* __restrict__ m, const int* __restrict__ src,
                          const int* __restrict__ dst, float* __restrict__ agg, int E)
{
    const long long tid = (long long)blockIdx.x * blockDim.x + threadIdx.x;
    const int e = (int)(tid >> 5);
    if (e >= E) return;
    const int j = ((int)tid & 31) << 2;       // lane -> 4 consecutive features
    const int s = src[e], d = dst[e];
    const float4 v = *(const float4*)&m[(size_t)s * HDIM + j];
    float* p = &agg[(size_t)d * HDIM + j];
    atomicAdd(p + 0, v.x);
    atomicAdd(p + 1, v.y);
    atomicAdd(p + 2, v.z);
    atomicAdd(p + 3, v.w);
}

// out[n,:] = in[n,:] * scale[n] + bias[:]
__global__ void k_scale_bias(float* __restrict__ out, const float* __restrict__ in,
                             const float* __restrict__ scale,
                             const float* __restrict__ bias, int N)
{
    const int i4 = blockIdx.x * blockDim.x + threadIdx.x;
    if (i4 >= N * 32) return;
    const int n = i4 >> 5;
    const int j = (i4 & 31) << 2;
    const float s = scale[n];
    const float4 v = *(const float4*)&in[(size_t)n * HDIM + j];
    float4 o;
    o.x = v.x * s + bias[j + 0];
    o.y = v.y * s + bias[j + 1];
    o.z = v.z * s + bias[j + 2];
    o.w = v.w * s + bias[j + 3];
    *(float4*)&out[(size_t)n * HDIM + j] = o;
}

// ---------------------------------------------------------------------------
// scores_r[n] = dot(T_r[n,:], w2); a = softmax over r (R=2);
// out[n,:] = a0*hs0[n,:] + a1*hs1[n,:].  One wave per node (wave32).
// ---------------------------------------------------------------------------
__launch_bounds__(256)
__global__ void k_attn_combine(const float* __restrict__ T0, const float* __restrict__ T1,
                               const float* __restrict__ w2,
                               const float* __restrict__ hs0, const float* __restrict__ hs1,
                               float* __restrict__ out, int N)
{
    const int node = blockIdx.x * 8 + (threadIdx.x >> 5);
    const int l = threadIdx.x & 31;
    if (node >= N) return;
    float s0 = 0.f, s1 = 0.f;
    #pragma unroll
    for (int q = l; q < QDIM; q += 32) {
        const float w = w2[q];
        s0 += T0[(size_t)node * QDIM + q] * w;
        s1 += T1[(size_t)node * QDIM + q] * w;
    }
    #pragma unroll
    for (int off = 16; off > 0; off >>= 1) {
        s0 += __shfl_down(s0, off);
        s1 += __shfl_down(s1, off);
    }
    s0 = __shfl(s0, 0);
    s1 = __shfl(s1, 0);
    const float mx = fmaxf(s0, s1);
    const float e0 = __expf(s0 - mx), e1 = __expf(s1 - mx);
    const float inv = 1.0f / (e0 + e1);
    const float a0 = e0 * inv, a1 = e1 * inv;
    const int h = l << 2;
    const float4 v0 = *(const float4*)&hs0[(size_t)node * HDIM + h];
    const float4 v1 = *(const float4*)&hs1[(size_t)node * HDIM + h];
    float4 o;
    o.x = a0 * v0.x + a1 * v1.x;
    o.y = a0 * v0.y + a1 * v1.y;
    o.z = a0 * v0.z + a1 * v1.z;
    o.w = a0 * v0.w + a1 * v1.w;
    *(float4*)&out[(size_t)node * HDIM + h] = o;
}

// ---------------------------------------------------------------------------
extern "C" void kernel_launch(void* const* d_in, const int* in_sizes, int n_in,
                              void* d_out, int out_size, void* d_ws, size_t ws_size,
                              hipStream_t stream)
{
    (void)in_sizes; (void)n_in; (void)out_size; (void)ws_size;

    const float* x_attr = (const float*)d_in[0];
    const float* x_stru = (const float*)d_in[1];
    const int*   e_attr = (const int*)d_in[2];
    const int*   e_stru = (const int*)d_in[3];
    const float* Wf     = (const float*)d_in[4];
    const float* convW  = (const float*)d_in[5];
    const float* convB  = (const float*)d_in[6];
    const float* aW1    = (const float*)d_in[7];
    const float* aB1    = (const float*)d_in[8];
    const float* aW2    = (const float*)d_in[9];
    const float* Wc     = (const float*)d_in[10];
    float* outp = (float*)d_out;
    float* ws   = (float*)d_ws;

    const size_t NH = (size_t)N_NODES * HDIM;   // 6.4M floats = 25.6 MB
    float* h0   = ws + 0 * NH;
    float* h1   = ws + 1 * NH;
    float* h2   = ws + 2 * NH;
    float* hA   = ws + 3 * NH;
    float* hs[2] = { ws + 4 * NH, ws + 5 * NH };
    float* mbuf = ws + 6 * NH;                       // also reused as T0/T1
    float* T0   = mbuf;
    float* T1   = mbuf + (size_t)N_NODES * QDIM;
    float* deg  = ws + 7 * NH;                       // 4*N: [r0_out, r0_in, r1_out, r1_in]

    const int gemmGrid = (N_NODES + 127) / 128;

    // Nc=128, C = A @ B^T  (h0 and final concat GEMMs)
    auto gemm_t = [&](const float* A, int lda, const float* B, int ldb,
                      const float* bias, float* C, int ldc, int K, int doTanh, int accum) {
        k_gemm_bf16<8, 1, 0><<<gemmGrid, 256, 0, stream>>>(
            A, lda, B, ldb, nullptr, bias, C, ldc, N_NODES, K, doTanh, accum);
    };
    // Nc=128, C = (rowScale * A) @ B  (GCN message GEMM)
    auto gemm_ns = [&](const float* A, int lda, const float* B, int ldb,
                       const float* rowScale, float* C, int ldc, int K) {
        k_gemm_bf16<8, 0, 1><<<gemmGrid, 256, 0, stream>>>(
            A, lda, B, ldb, rowScale, nullptr, C, ldc, N_NODES, K, 0, 0);
    };
    // Nc=64, C = tanh(A @ B + bias)  (attention projection)
    auto gemm_q = [&](const float* A, int lda, const float* B, int ldb,
                      const float* bias, float* C, int ldc, int K) {
        k_gemm_bf16<4, 0, 0><<<gemmGrid, 256, 0, stream>>>(
            A, lda, B, ldb, nullptr, bias, C, ldc, N_NODES, K, 1, 0);
    };

    struct Run { int sign; const float* x; const int* ed; float* out; };
    const Run runs[4] = {
        {0, x_attr, e_attr, outp + 0 * NH},
        {0, x_stru, e_stru, outp + 1 * NH},
        {1, x_attr, e_attr, outp + 2 * NH},
        {1, x_stru, e_stru, outp + 3 * NH},
    };

    for (int ri = 0; ri < 4; ++ri) {
        const int sg = runs[ri].sign;
        const int* src[2]; const int* dst[2];
        for (int r = 0; r < RREL; ++r) {
            src[r] = runs[ri].ed + ((size_t)(sg * RREL + r) * 2 + 0) * EDGES;
            dst[r] = runs[ri].ed + ((size_t)(sg * RREL + r) * 2 + 1) * EDGES;
        }

        // degrees (shared by all 3 hetero layers of this run)
        hipMemsetAsync(deg, 0, 4 * N_NODES * sizeof(float), stream);
        for (int r = 0; r < RREL; ++r)
            k_degree<<<(EDGES + 255) / 256, 256, 0, stream>>>(
                src[r], dst[r], deg + 2 * r * N_NODES, deg + (2 * r + 1) * N_NODES, EDGES);
        k_rsqrt_clamp<<<(4 * N_NODES + 255) / 256, 256, 0, stream>>>(deg, 4 * N_NODES);

        // h0 = tanh(x @ Wf[sg]^T)
        gemm_t(runs[ri].x, FDIM, Wf + (size_t)sg * HDIM * FDIM, FDIM,
               nullptr, h0, HDIM, FDIM, /*tanh=*/1, 0);

        auto het = [&](const float* hin, int l, float* outb) {
            for (int r = 0; r < RREL; ++r) {
                const float* Wl = convW + (size_t)((sg * LDEP + l) * RREL + r) * HDIM * HDIM;
                const float* bl = convB + (size_t)((sg * LDEP + l) * RREL + r) * HDIM;
                // m = (h * deg_out^-1/2) @ W
                gemm_ns(hin, HDIM, Wl, HDIM, deg + 2 * r * N_NODES, mbuf, HDIM, HDIM);
                // agg = segment_sum(m[src], dst)
                hipMemsetAsync(hs[r], 0, NH * sizeof(float), stream);
                k_scatter<<<(EDGES * 32 + 255) / 256, 256, 0, stream>>>(
                    mbuf, src[r], dst[r], hs[r], EDGES);
                // hs[r] = agg * deg_in^-1/2 + b
                k_scale_bias<<<(N_NODES * 32 + 255) / 256, 256, 0, stream>>>(
                    hs[r], hs[r], deg + (2 * r + 1) * N_NODES, bl, N_NODES);
            }
            const float* W1 = aW1 + (size_t)(sg * LDEP + l) * HDIM * QDIM;
            const float* b1 = aB1 + (size_t)(sg * LDEP + l) * QDIM;
            const float* w2 = aW2 + (size_t)(sg * LDEP + l) * QDIM;
            // T_r = tanh(hs[r] @ W1 + b1)
            gemm_q(hs[0], HDIM, W1, QDIM, b1, T0, QDIM, HDIM);
            gemm_q(hs[1], HDIM, W1, QDIM, b1, T1, QDIM, HDIM);
            k_attn_combine<<<(N_NODES + 7) / 8, 256, 0, stream>>>(
                T0, T1, w2, hs[0], hs[1], outb, N_NODES);
        };

        het(h0, 1, h1);   // 1-hop via last conv layer
        het(h0, 0, hA);
        het(hA, 1, h2);   // 2-hop

        // y = concat(h0,h1,h2) @ Wc^T  (three accumulated K=128 GEMMs)
        const float* Wcb = Wc + (size_t)sg * HDIM * 3 * HDIM;
        gemm_t(h0, HDIM, Wcb + 0,      3 * HDIM, nullptr, runs[ri].out, HDIM, HDIM, 0, 0);
        gemm_t(h1, HDIM, Wcb + HDIM,   3 * HDIM, nullptr, runs[ri].out, HDIM, HDIM, 0, 1);
        gemm_t(h2, HDIM, Wcb + 2*HDIM, 3 * HDIM, nullptr, runs[ri].out, HDIM, HDIM, 0, 1);
    }
}
